// NeuralCache_83365315215900
// MI455X (gfx1250) — matrix operand order repaired and checked by reading, not compiled, not run on hardware
//
#include <hip/hip_runtime.h>

#define HASH_BITS   16
#define NUM_TABLES  4
#define INPUT_DIM   1024
#define OUTPUT_DIM  1024
#define RAM         65536
#define BATCH       4096
#define KCHUNKS     (INPUT_DIM / 32)    // 32 K-chunks of 32

typedef __attribute__((ext_vector_type(16))) __bf16 v16bf;
typedef __attribute__((ext_vector_type(8)))  float  v8f;

// bf16 (raw bits) -> f32
static __device__ __forceinline__ float bf_lo(unsigned u) {
    return __builtin_bit_cast(float, u << 16);
}
static __device__ __forceinline__ float bf_hi(unsigned u) {
    return __builtin_bit_cast(float, u & 0xffff0000u);
}

// ---------------------------------------------------------------------------
// Kernel 0: pre-pack planes (f32, [t][K][h]) into bf16 WMMA B-fragments laid
// out exactly as registers want them: fragment (c,t) = 32 lanes x 16 bf16,
// lane<16 holds column N=lane, K = c*32 + {0..15}; lane>=16 holds K+16.
// Interleaved [c][t] so kernel 1 streams 4KB contiguously per K-iteration.
// Total: 128 KB, packed once, L2-resident thereafter.
// ---------------------------------------------------------------------------
__global__ __launch_bounds__(128) void pack_planes_kernel(
    const float* __restrict__ planes, __bf16* __restrict__ packedB)
{
    const int lane  = threadIdx.x & 31;
    const int wg    = blockIdx.x * 4 + (threadIdx.x >> 5);  // 0..127
    const int c     = wg >> 2;          // K-chunk
    const int t     = wg & 3;           // table
    const int khalf = lane >> 4;

    const float* pb = planes + (size_t)t * (INPUT_DIM * HASH_BITS) + (lane & 15);
    v16bf frag;
#pragma unroll
    for (int j = 0; j < 16; ++j) {
        const int K = c * 32 + khalf * 16 + j;
        frag[j] = (__bf16)pb[(size_t)K * HASH_BITS];
    }
    *reinterpret_cast<v16bf*>(packedB + (size_t)wg * 512 + lane * 16) = frag;
}

// ---------------------------------------------------------------------------
// Kernel 1: LSH projection via WMMA + sign-bit hash via ballot + row norms.
// One wave per 16-row batch tile; 4 accumulators (one per table).
// C/D layout: VGPR r holds rows {r, r+8}, hash bit h = lane (mod 16), so
// ballot(proj>0) directly yields the 16-bit LSH address per row.
// ---------------------------------------------------------------------------
__global__ __launch_bounds__(128) void hash_wmma_kernel(
    const float* __restrict__ x, const __bf16* __restrict__ packedB,
    int* __restrict__ addresses, float* __restrict__ invnorm)
{
    const int lane    = threadIdx.x & 31;
    const int wave    = threadIdx.x >> 5;
    const int tile    = blockIdx.x * 4 + wave;     // 256 tiles of 16 rows
    const int rowbase = tile * 16;
    const int row     = rowbase + (lane & 15);
    const int khalf   = lane >> 4;                 // 0 or 1

    v8f acc[NUM_TABLES] = {};
    float ss = 0.0f;

    const float*  xr = x + (size_t)row * INPUT_DIM;
    const __bf16* pB = packedB + (size_t)lane * 16;   // + (c*4+t)*512

    for (int c = 0; c < KCHUNKS; ++c) {
        const int kb = c * 32;
        // ---- A fragment (16x32 bf16): lane<16 K={kb..kb+15},
        //      lane>=16 shifted by 16. Lanes L and L+16 share a row and read
        //      complementary halves of one 128B line -> 100% line use. ----
        const int k0 = kb + khalf * 16;
        float4 f0 = *reinterpret_cast<const float4*>(xr + k0);
        float4 f1 = *reinterpret_cast<const float4*>(xr + k0 + 4);
        float4 f2 = *reinterpret_cast<const float4*>(xr + k0 + 8);
        float4 f3 = *reinterpret_cast<const float4*>(xr + k0 + 12);
        float xs[16] = { f0.x,f0.y,f0.z,f0.w, f1.x,f1.y,f1.z,f1.w,
                         f2.x,f2.y,f2.z,f2.w, f3.x,f3.y,f3.z,f3.w };
        v16bf a;
#pragma unroll
        for (int j = 0; j < 16; ++j) {
            ss   = fmaf(xs[j], xs[j], ss);   // row-norm accumulation for free
            a[j] = (__bf16)xs[j];
        }
        // ---- B fragments: one 32B coalesced load per table (pre-packed) ----
#pragma unroll
        for (int t = 0; t < NUM_TABLES; ++t) {
            const v16bf bfrag = *reinterpret_cast<const v16bf*>(
                pB + (size_t)(c * NUM_TABLES + t) * 512);
            acc[t] = __builtin_amdgcn_wmma_f32_16x16x32_bf16(
                /*neg_a=*/false, a, /*neg_b=*/false, bfrag,
                /*c_mod=*/(short)0, acc[t], /*reuse_a=*/false, /*reuse_b=*/false);
        }
    }

    // Combine complementary K halves of each row's sum of squares.
    ss += __shfl_xor(ss, 16, 32);

    // Sign-bit hash: ballot over each accumulator VGPR gives 16 bits for row r
    // (low half) and row r+8 (high half), bit position == hash bit.
#pragma unroll
    for (int t = 0; t < NUM_TABLES; ++t) {
        unsigned myaddr = 0;
#pragma unroll
        for (int r = 0; r < 8; ++r) {
            const bool pred = (acc[t][r] > 0.0f);
#if __has_builtin(__builtin_amdgcn_ballot_w32)
            const unsigned m = __builtin_amdgcn_ballot_w32(pred);
#else
            const unsigned m = (unsigned)__ballot((int)pred);
#endif
            if ((lane & 7) == r) myaddr = (lane < 8) ? (m & 0xffffu) : (m >> 16);
        }
        if (lane < 16)
            addresses[(size_t)(rowbase + lane) * NUM_TABLES + t] = (int)myaddr;
    }
    if (lane < 16)
        invnorm[rowbase + lane] = rsqrtf(fmaxf(ss, 1e-24f));
}

// ---------------------------------------------------------------------------
// Kernel 2: hash-indexed gather + cosine sims + blended selection + values out.
// One block (4 waves) per batch element; wave t handles table t.
// ---------------------------------------------------------------------------
__global__ __launch_bounds__(128) void gather_select_kernel(
    const float* __restrict__ x,
    const unsigned short* __restrict__ keys,     // bf16 raw bits
    const unsigned short* __restrict__ values,   // bf16 raw bits
    const unsigned char* __restrict__ valid,
    const float* __restrict__ reliability,
    const float* __restrict__ last_access,
    const int* __restrict__ addresses,
    const float* __restrict__ invnorm,
    float* __restrict__ out)
{
    const int b    = blockIdx.x;
    const int tid  = threadIdx.x;
    const int t    = tid >> 5;       // wave id == table id
    const int lane = tid & 31;

    __shared__ float s_sim[NUM_TABLES], s_rel[NUM_TABLES], s_rec[NUM_TABLES];
    __shared__ int   s_addr[NUM_TABLES], s_val[NUM_TABLES];
    __shared__ int   s_hit, s_opb, s_opa;

    const int    addr = addresses[(size_t)b * NUM_TABLES + t];
    const size_t slot = (size_t)t * RAM + (unsigned)addr;

    // Warm L2 with the candidate values row (global_prefetch_b8): 32 lanes x
    // 64B covers the whole 2KB bf16 row before we know which table wins.
    __builtin_prefetch((const char*)(values + slot * OUTPUT_DIM) + lane * 64, 0, 1);

    const unsigned short* krow = keys + slot * INPUT_DIM;
    const float*          xr   = x + (size_t)b * INPUT_DIM;

    float sum = 0.0f;
#pragma unroll
    for (int c = 0; c < 4; ++c) {
        const int i = c * 256 + lane * 8;            // 8 bf16 per lane per iter
        uint4  kv = *reinterpret_cast<const uint4*>(krow + i);
        float4 x0 = *reinterpret_cast<const float4*>(xr + i);
        float4 x1 = *reinterpret_cast<const float4*>(xr + i + 4);
        sum = fmaf(x0.x, bf_lo(kv.x), sum);
        sum = fmaf(x0.y, bf_hi(kv.x), sum);
        sum = fmaf(x0.z, bf_lo(kv.y), sum);
        sum = fmaf(x0.w, bf_hi(kv.y), sum);
        sum = fmaf(x1.x, bf_lo(kv.z), sum);
        sum = fmaf(x1.y, bf_hi(kv.z), sum);
        sum = fmaf(x1.z, bf_lo(kv.w), sum);
        sum = fmaf(x1.w, bf_hi(kv.w), sum);
    }
#pragma unroll
    for (int off = 16; off > 0; off >>= 1)
        sum += __shfl_down(sum, off, 32);

    if (lane == 0) {
        s_sim[t]  = sum * invnorm[b];
        s_addr[t] = addr;
        s_val[t]  = valid[slot];
        s_rel[t]  = reliability[slot];
        const float age = fmaxf(1.0f - last_access[slot], 0.0f);  // NOW = 1.0
        s_rec[t]  = 1.0f / (1.0f + age);
    }
    __syncthreads();

    if (tid == 0) {
        float best_blend  = -INFINITY; int best_table = 0;
        float best_masked = -INFINITY; int op_best    = 0;
        int any = 0;
#pragma unroll
        for (int k = 0; k < NUM_TABLES; ++k) {
            const float sim = s_sim[k];
            const int   hit = (s_val[k] != 0) && (sim >= 0.0f);
            any |= hit;
            const float sm = hit ? sim : -INFINITY;
            if (sm > best_masked) { best_masked = sm; op_best = k; }   // first-max
            const float bl = 0.7f * sim + 0.2f * s_rel[k] + 0.1f * s_rec[k];
            if (bl > best_blend) { best_blend = bl; best_table = k; }  // first-max
        }
        float* o = out + (size_t)BATCH * OUTPUT_DIM;
        o[b]             = any ? 1.0f : 0.0f;                              // hit_mask
        o[BATCH + b]     = any ? s_sim[best_table] : 0.0f;                 // max_sim
        o[2 * BATCH + b] = (float)(any ? s_addr[best_table] : s_addr[0]);  // hit_addrs
        o[3 * BATCH + b] = (float)(any ? best_table : 0);                  // hit_tables
        s_hit = any; s_opb = op_best; s_opa = s_addr[op_best];
    }
    __syncthreads();

    // Stream the winning values row (or zeros) — 8 elements per thread.
    const int any = s_hit;
    const unsigned short* vrow =
        values + ((size_t)s_opb * RAM + (unsigned)s_opa) * OUTPUT_DIM;
    const int i = tid * 8;
    float4 r0 = make_float4(0.f, 0.f, 0.f, 0.f), r1 = r0;
    if (any) {
        uint4 v = *reinterpret_cast<const uint4*>(vrow + i);
        r0 = make_float4(bf_lo(v.x), bf_hi(v.x), bf_lo(v.y), bf_hi(v.y));
        r1 = make_float4(bf_lo(v.z), bf_hi(v.z), bf_lo(v.w), bf_hi(v.w));
    }
    float* orow = out + (size_t)b * OUTPUT_DIM + i;
    *reinterpret_cast<float4*>(orow)     = r0;
    *reinterpret_cast<float4*>(orow + 4) = r1;
}

extern "C" void kernel_launch(void* const* d_in, const int* in_sizes, int n_in,
                              void* d_out, int out_size, void* d_ws, size_t ws_size,
                              hipStream_t stream) {
    const float*          x           = (const float*)d_in[0];
    const float*          planes      = (const float*)d_in[1];
    const unsigned short* keys        = (const unsigned short*)d_in[2];  // bf16
    const unsigned short* values      = (const unsigned short*)d_in[3];  // bf16
    const unsigned char*  valid       = (const unsigned char*)d_in[4];
    const float*          reliability = (const float*)d_in[5];
    const float*          last_access = (const float*)d_in[6];
    float* out = (float*)d_out;

    // Workspace layout: addresses [B*T] int (64KB) | invnorm [B] f32 (16KB)
    //                 | packedB 4*32*512 bf16 (128KB)
    int*    addresses = (int*)d_ws;
    float*  invnorm   = (float*)((char*)d_ws + (size_t)BATCH * NUM_TABLES * sizeof(int));
    __bf16* packedB   = (__bf16*)((char*)d_ws + (size_t)BATCH * NUM_TABLES * sizeof(int)
                                             + (size_t)BATCH * sizeof(float));

    // 128 fragments / 4 waves per block = 32 blocks
    pack_planes_kernel<<<dim3(32), dim3(128), 0, stream>>>(planes, packedB);

    // 256 row-tiles / 4 waves per block = 64 blocks
    hash_wmma_kernel<<<dim3(BATCH / 64), dim3(128), 0, stream>>>(
        x, packedB, addresses, invnorm);

    gather_select_kernel<<<dim3(BATCH), dim3(128), 0, stream>>>(
        x, keys, values, valid, reliability, last_access, addresses, invnorm, out);
}
